// MOF_Net3_41240275976363
// MI455X (gfx1250) — compile-verified, matrix operand order checked by reading, not run on hardware
//
#include <hip/hip_runtime.h>

typedef _Float16 f16;
typedef __attribute__((ext_vector_type(16))) _Float16 v16h;
typedef __attribute__((ext_vector_type(2)))  _Float16 f16x2;
typedef __attribute__((ext_vector_type(8)))  float    v8f;

#define D_NODE 16
#define D_EDGE 8
#define D_IN   40     // 2*D_NODE + D_EDGE
#define D_HID  64
#define TILE_M 16     // edges per wave-tile (WMMA N in transposed form)
#define NWAVES 8
#define BLOCK  (NWAVES * 32)
#define NG_LDS 512    // LDS graph-accumulator bins

// K position held by (vgpr-pair v, lane-half hi) for 16-bit WMMA operands
// (ISA 7.12.2: v0..3 -> K 0..7, v4..7 -> K 16..23, upper lane half +8).
static __device__ __forceinline__ int kmap(int v, int hi) {
  return ((v < 4) ? (2 * v) : (8 + 2 * v)) + (hi ? 8 : 0);
}

// A operand rows from an LDS row-major [row][64] f16 matrix; K pairs are
// contiguous so the compiler merges these into ds_load_b128.
static __device__ __forceinline__ v16h load_A_rows(const f16* Wt, int rowidx,
                                                   int c, int hi) {
  const f16* row = Wt + rowidx * 64;
  v16h a;
#pragma unroll
  for (int v = 0; v < 8; ++v) {
    int k = 32 * c + kmap(v, hi);
    a[2 * v]     = row[k];
    a[2 * v + 1] = row[k + 1];
  }
  return a;
}

static __device__ __forceinline__ f16x2 pk(float x, float y) {
  return __builtin_bit_cast(f16x2, __builtin_amdgcn_cvt_pkrtz(x, y));
}

// D layout of layer L == B layout of layer L+1 (lane: edge column n,
// VGPR r: hidden 16*mt + r + 8*hi). Pack two accumulators, then relu in
// f16 on the packed vector: 8x v_pk_max_num_f16 instead of 16x f32 max.
// (cvt_pk_rtz is monotonic & sign-preserving, so relu commutes with it.)
static __device__ __forceinline__ v16h pack_relu(v8f a, v8f b) {
  union { v16h v; f16x2 p[8]; } u;
#pragma unroll
  for (int j = 0; j < 4; ++j) {
    u.p[j]     = pk(a[2 * j], a[2 * j + 1]);
    u.p[4 + j] = pk(b[2 * j], b[2 * j + 1]);
  }
  v16h zero = {};
  return __builtin_elementwise_max(u.v, zero);
}

__global__ void __launch_bounds__(BLOCK) zero_out_kernel(float* out, int n) {
  int i = blockIdx.x * blockDim.x + threadIdx.x;
  if (i < n) out[i] = 0.f;
}

__global__ void __launch_bounds__(BLOCK)
molgcn_kernel(const float* __restrict__ x,
              const float* __restrict__ edge_attr,
              const float* __restrict__ W1, const float* __restrict__ b1,
              const float* __restrict__ W2, const float* __restrict__ b2,
              const float* __restrict__ W3, const float* __restrict__ b3,
              const int*   __restrict__ eidx,   // [2*E]: row0 = src, row1 = dst
              const int*   __restrict__ batch,  // [N]
              float*       __restrict__ out,    // [G]
              int E, int G)
{
  // Transposed weights in LDS: WtT[h*64 + k] = W[k][h]; W1 gets K-pad rows
  // 40..62 = 0 and k==63 = b1[h] (bias fold, feature col 63 carries 1.0).
  __shared__ f16   W1sT[64 * 64];
  __shared__ f16   W2sT[64 * 64];
  __shared__ float gacc[NG_LDS];

  const int tid  = threadIdx.x;
  const int lane = tid & 31;
  const int wave = tid >> 5;
  const int hi   = (lane >> 4) & 1;
  const int nloc = lane & 15;

  for (int i = tid; i < 64 * 64; i += BLOCK) {
    int h = i >> 6, k = i & 63;
    f16 w1v = (f16)0.f;
    if (k < D_IN)     w1v = (f16)W1[k * 64 + h];
    else if (k == 63) w1v = (f16)b1[h];
    W1sT[i] = w1v;
    W2sT[i] = (f16)W2[k * 64 + h];
  }
  for (int i = tid; i < NG_LDS; i += BLOCK) gacc[i] = 0.f;
  __syncthreads();

  // ---- loop-invariant registers ----
  v16h A1r[2][4], A2r[2][4];                 // W1^T, W2^T tiles
#pragma unroll
  for (int c = 0; c < 2; ++c)
#pragma unroll
    for (int mt = 0; mt < 4; ++mt) {
      A1r[c][mt] = load_A_rows(W1sT, 16 * mt + nloc, c, hi);
      A2r[c][mt] = load_A_rows(W2sT, 16 * mt + nloc, c, hi);
    }
  v16h A3r[2];                               // w3 as row 0 of a 16x64 A tile
#pragma unroll
  for (int c = 0; c < 2; ++c) {
#pragma unroll
    for (int v = 0; v < 8; ++v) {
      int k = 32 * c + kmap(v, hi);
      A3r[c][2 * v]     = (nloc == 0) ? (f16)W3[k]     : (f16)0.f;
      A3r[c][2 * v + 1] = (nloc == 0) ? (f16)W3[k + 1] : (f16)0.f;
    }
  }
  v8f b2init[4];                             // layer-2 bias rides in as C
#pragma unroll
  for (int mt = 0; mt < 4; ++mt)
#pragma unroll
    for (int r = 0; r < 8; ++r)
      b2init[mt][r] = b2[16 * mt + r + 8 * hi];
  v16h Bc1_hi1 = {};                         // feature chunk1, upper lane half:
  Bc1_hi1[15] = (f16)1.0f;                   // only K=63 carries the bias 1.0
  const float b3v = b3[0];

  const int dst_base = E;
  const int ntiles = (E + TILE_M - 1) / TILE_M;
  const int gwave  = blockIdx.x * NWAVES + wave;
  const int nw     = gridDim.x * NWAVES;

  for (int tile = gwave; tile < ntiles; tile += nw) {
    const int ebase = tile * TILE_M;
    const int e = ebase + nloc;              // this lane's edge column

    // ---- build feature B operand straight from global (no LDS) ----
    int di = 0;
    float4 d0 = {}, d1 = {}, s0 = {}, s1 = {};
    if (e < E) {
      di = eidx[dst_base + e];
      int si = eidx[e];
      const float4* dp = (const float4*)(x + (long)di * D_NODE + 8 * hi);
      const float4* sp = (const float4*)(x + (long)si * D_NODE + 8 * hi);
      d0 = dp[0]; d1 = dp[1];
      s0 = sp[0]; s1 = sp[1];
    }
    v16h Bc0;                                // K: x[dst] then x[src] halves
    {
      union { v16h v; f16x2 p[8]; } u;
      u.p[0] = pk(d0.x, d0.y); u.p[1] = pk(d0.z, d0.w);
      u.p[2] = pk(d1.x, d1.y); u.p[3] = pk(d1.z, d1.w);
      u.p[4] = pk(s0.x, s0.y); u.p[5] = pk(s0.z, s0.w);
      u.p[6] = pk(s1.x, s1.y); u.p[7] = pk(s1.z, s1.w);
      Bc0 = u.v;
    }
    v16h Bc1;
    if (hi == 0) {                           // K 32..39 = edge_attr, rest 0
      float4 a0 = {}, a1 = {};
      if (e < E) {
        const float4* ap = (const float4*)(edge_attr + (long)e * D_EDGE);
        a0 = ap[0]; a1 = ap[1];
      }
      union { v16h v; f16x2 p[8]; } u;
      u.p[0] = pk(a0.x, a0.y); u.p[1] = pk(a0.z, a0.w);
      u.p[2] = pk(a1.x, a1.y); u.p[3] = pk(a1.z, a1.w);
      u.p[4] = f16x2{}; u.p[5] = f16x2{}; u.p[6] = f16x2{}; u.p[7] = f16x2{};
      Bc1 = u.v;
    } else {
      Bc1 = Bc1_hi1;
    }

    // ---- layer 1: z1^T = W1^T @ m^T (bias folded via K=63) ----
    v8f acc1[4];
#pragma unroll
    for (int mt = 0; mt < 4; ++mt) {
      v8f z = {};
      acc1[mt] = __builtin_amdgcn_wmma_f32_16x16x32_f16(false, A1r[0][mt], false, Bc0,
                                                        (short)0, z, false, false);
      acc1[mt] = __builtin_amdgcn_wmma_f32_16x16x32_f16(false, A1r[1][mt], false, Bc1,
                                                        (short)0, acc1[mt], false, false);
    }

    // ---- layer 2: D-of-layer1 is already B layout; pack + pk-relu in-lane ----
    v16h B2c0 = pack_relu(acc1[0], acc1[1]);
    v16h B2c1 = pack_relu(acc1[2], acc1[3]);
    v8f acc2[4];
#pragma unroll
    for (int mt = 0; mt < 4; ++mt) {
      acc2[mt] = __builtin_amdgcn_wmma_f32_16x16x32_f16(false, A2r[0][mt], false, B2c0,
                                                        (short)0, b2init[mt], false, false);
      acc2[mt] = __builtin_amdgcn_wmma_f32_16x16x32_f16(false, A2r[1][mt], false, B2c1,
                                                        (short)0, acc2[mt], false, false);
    }

    // ---- layer 3: msg^T = w3^T @ relu(h2^T) as 2 WMMAs ----
    v16h B3c0 = pack_relu(acc2[0], acc2[1]);
    v16h B3c1 = pack_relu(acc2[2], acc2[3]);
    v8f z3 = {};
    v8f acc3 = __builtin_amdgcn_wmma_f32_16x16x32_f16(false, A3r[0], false, B3c0,
                                                      (short)0, z3, false, false);
    acc3 = __builtin_amdgcn_wmma_f32_16x16x32_f16(false, A3r[1], false, B3c1,
                                                  (short)0, acc3, false, false);

    // msg for edge column nloc sits in row 0: VGPR 0 of lanes 0..15
    if (hi == 0 && e < E) {
      float msg = acc3[0] + b3v;
      int g = batch[di];
      if (g < NG_LDS) atomicAdd(&gacc[g], msg);        // LDS bin (500 graphs fit)
      else            atomicAdd(&out[g], msg * 0.5f);  // safety fallback
    }
  }

  // ---- flush per-block graph bins: global atomics, fused with /2 ----
  __syncthreads();
  int glim = (G < NG_LDS) ? G : NG_LDS;
  for (int i = tid; i < glim; i += BLOCK) {
    float v = gacc[i];
    if (v != 0.f) atomicAdd(&out[i], v * 0.5f);
  }
}

extern "C" void kernel_launch(void* const* d_in, const int* in_sizes, int n_in,
                              void* d_out, int out_size, void* d_ws, size_t ws_size,
                              hipStream_t stream) {
  const float* x   = (const float*)d_in[0];
  const float* ea  = (const float*)d_in[1];
  const float* W1  = (const float*)d_in[2];
  const float* b1  = (const float*)d_in[3];
  const float* W2  = (const float*)d_in[4];
  const float* b2  = (const float*)d_in[5];
  const float* W3  = (const float*)d_in[6];
  const float* b3  = (const float*)d_in[7];
  const int*   ei  = (const int*)d_in[8];   // (2, E): row0 src, row1 dst
  const int*   bat = (const int*)d_in[9];
  float* out = (float*)d_out;

  const int E = in_sizes[8] / 2;
  const int G = out_size;

  zero_out_kernel<<<(G + BLOCK - 1) / BLOCK, BLOCK, 0, stream>>>(out, G);

  const int blocks = 512;   // persistent waves; 100k tiles / 4096 waves ~ 24 tiles each
  molgcn_kernel<<<blocks, BLOCK, 0, stream>>>(x, ea, W1, b1, W2, b2, W3, b3,
                                              ei, bat, out, E, G);
}
